// GAT_Light_37263136260509
// MI455X (gfx1250) — matrix operand order repaired and checked by reading, not compiled, not run on hardware
//
#include <hip/hip_runtime.h>
#include <math.h>

#define NEG_SLOPE 0.2f
#define GAT_EPS   1e-16f

typedef __attribute__((ext_vector_type(16))) _Float16 v16h;
typedef __attribute__((ext_vector_type(8)))  float    v8f;

__device__ __forceinline__ float leakyr(float e) {
    return e > 0.0f ? e : NEG_SLOPE * e;
}

// float atomic-max via signed/unsigned bit-order trick (init cell to -inf)
__device__ __forceinline__ void atomicMaxF(float* addr, float v) {
    if (v >= 0.0f) atomicMax((int*)addr, __float_as_int(v));
    else           atomicMin((unsigned int*)addr, __float_as_uint(v));
}

// ---------------- init all accumulators (re-run every call: graph-safe) ----
__global__ void gat_init_ws(float* m1, float* d1, float* out1,
                            float* m2, float* d2, float* out2,
                            float* pooled, int N) {
    int i = blockIdx.x * blockDim.x + threadIdx.x;
    const float ninf = __int_as_float(0xff800000);
    if (i < N * 64) out1[i] = 0.0f;
    if (i < N * 16) out2[i] = 0.0f;
    if (i < N * 4)  { m1[i] = ninf; d1[i] = 0.0f; }
    if (i < N * 2)  { m2[i] = ninf; d2[i] = 0.0f; }
    if (i < 16)     pooled[i] = 0.0f;
}

// ---------------- layer 1 node phase: h1 = x @ W1 (K=1), alpha dots --------
__global__ void gat_node1(const float* __restrict__ x, const float* __restrict__ W1,
                          const float* __restrict__ a_src1, const float* __restrict__ a_dst1,
                          float* __restrict__ h1, float* __restrict__ as1,
                          float* __restrict__ ad1, int N) {
    int i = blockIdx.x * blockDim.x + threadIdx.x;
    if (i >= N) return;
    float xv = x[i];
    float s[4] = {0.f, 0.f, 0.f, 0.f};
    float d[4] = {0.f, 0.f, 0.f, 0.f};
    #pragma unroll
    for (int j = 0; j < 64; ++j) {
        float h = xv * W1[j];
        h1[(size_t)i * 64 + j] = h;
        int hd = j >> 4;
        s[hd] += h * a_src1[j];
        d[hd] += h * a_dst1[j];
    }
    #pragma unroll
    for (int hd = 0; hd < 4; ++hd) {
        as1[i * 4 + hd] = s[hd];
        ad1[i * 4 + hd] = d[hd];
    }
}

// ---------------- edge pass A: segment max of leaky(e) over dst ------------
template <int H>
__global__ void gat_edge_max(const int* __restrict__ ei, int E, int N,
                             const float* __restrict__ as, const float* __restrict__ ad,
                             float* __restrict__ m) {
    int t = blockIdx.x * blockDim.x + threadIdx.x;
    int Et = E + N;
    if (t >= Et) return;
    int s, d;
    if (t < E) { s = ei[t]; d = ei[E + t]; } else { s = t - E; d = s; }
    #pragma unroll
    for (int h = 0; h < H; ++h) {
        float e = leakyr(as[s * H + h] + ad[d * H + h]);
        atomicMaxF(&m[d * H + h], e);
    }
}

// ---------------- edge pass B: denom = segment sum of exp(e - m[dst]) ------
template <int H>
__global__ void gat_edge_sum(const int* __restrict__ ei, int E, int N,
                             const float* __restrict__ as, const float* __restrict__ ad,
                             const float* __restrict__ m, float* __restrict__ den) {
    int t = blockIdx.x * blockDim.x + threadIdx.x;
    int Et = E + N;
    if (t >= Et) return;
    int s, d;
    if (t < E) { s = ei[t]; d = ei[E + t]; } else { s = t - E; d = s; }
    #pragma unroll
    for (int h = 0; h < H; ++h) {
        float e = leakyr(as[s * H + h] + ad[d * H + h]);
        atomicAdd(&den[d * H + h], __expf(e - m[d * H + h]));
    }
}

// ---------------- edge pass C: out[dst] += alpha * h[src] ------------------
// one thread per (edge, head)
template <int H, int C>
__global__ void gat_edge_agg(const int* __restrict__ ei, int E, int N,
                             const float* __restrict__ as, const float* __restrict__ ad,
                             const float* __restrict__ m, const float* __restrict__ den,
                             const float* __restrict__ hsrc, float* __restrict__ out) {
    int t = blockIdx.x * blockDim.x + threadIdx.x;
    int Et = E + N;
    if (t >= Et * H) return;
    int eidx = t / H;
    int h = t - eidx * H;
    int s, d;
    if (eidx < E) { s = ei[eidx]; d = ei[E + eidx]; } else { s = eidx - E; d = s; }
    float e = leakyr(as[s * H + h] + ad[d * H + h]);
    float alpha = __expf(e - m[d * H + h]) / (den[d * H + h] + GAT_EPS);
    const float* hp = hsrc + (size_t)s * (H * C) + h * C;
    float* op = out + (size_t)d * (H * C) + h * C;
    #pragma unroll
    for (int c = 0; c < C; ++c) atomicAdd(&op[c], alpha * hp[c]);
}

// ---------------- bias + relu for layer-1 output (in place) ----------------
__global__ void gat_bias_relu64(float* __restrict__ o, const float* __restrict__ b, int N) {
    int i = blockIdx.x * blockDim.x + threadIdx.x;
    if (i < N * 64) o[i] = fmaxf(o[i] + b[i & 63], 0.0f);
}

// ---------------- layer-2 GEMM via WMMA: h2[N,16] = x2[N,64] @ W2[64,16] ---
// One wave32 per FULL 16-row tile (grid covers only full tiles -> branch-free
// stores; tail rows handled by gat_gemm_tail).
// A layout (16-bit A 16x32): element e -> col K = 16*half + e, row = lane&15
// B layout (16-bit B 32x16): element e -> row K = 16*half + e, col = lane&15
// D layout: VGPR r -> M = r + 8*half, N = lane&15
__global__ __launch_bounds__(32) void gat_gemm_h2(const float* __restrict__ x2,
                                                  const float* __restrict__ W2,
                                                  float* __restrict__ h2) {
    int lane = threadIdx.x;
    int half = lane >> 4;
    int l = lane & 15;
    int rowbase = blockIdx.x * 16;
    const float* ap = x2 + (size_t)(rowbase + l) * 64;

    v8f c = {};
    #pragma unroll
    for (int kb = 0; kb < 2; ++kb) {
        v16h a, b;
        #pragma unroll
        for (int e = 0; e < 16; ++e) {
            int ka = kb * 32 + half * 16 + e;
            a[e] = (_Float16)ap[ka];
            int kr = kb * 32 + half * 16 + e;
            b[e] = (_Float16)W2[kr * 16 + l];
        }
        c = __builtin_amdgcn_wmma_f32_16x16x32_f16(
                /*neg_a=*/false, a, /*neg_b=*/false, b,
                /*c_mod=*/(short)0, c, /*reuse_a=*/false, /*reuse_b=*/false);
    }
    #pragma unroll
    for (int r = 0; r < 8; ++r) {
        h2[(size_t)(rowbase + r + half * 8) * 16 + l] = c[r];
    }
}

// scalar tail for rows [start, N) when N % 16 != 0
__global__ void gat_gemm_tail(const float* __restrict__ x2, const float* __restrict__ W2,
                              float* __restrict__ h2, int start, int N) {
    int i = start + blockIdx.x * blockDim.x + threadIdx.x;
    if (i >= N) return;
    #pragma unroll
    for (int j = 0; j < 16; ++j) {
        float acc = 0.0f;
        for (int k = 0; k < 64; ++k)
            acc += (float)((_Float16)x2[(size_t)i * 64 + k]) *
                   (float)((_Float16)W2[k * 16 + j]);
        h2[(size_t)i * 16 + j] = acc;
    }
}

// ---------------- layer-2 alpha dots ---------------------------------------
__global__ void gat_node2_alpha(const float* __restrict__ h2,
                                const float* __restrict__ a_src2,
                                const float* __restrict__ a_dst2,
                                float* __restrict__ as2, float* __restrict__ ad2, int N) {
    int i = blockIdx.x * blockDim.x + threadIdx.x;
    if (i >= N) return;
    #pragma unroll
    for (int hd = 0; hd < 2; ++hd) {
        float s = 0.f, d = 0.f;
        #pragma unroll
        for (int c = 0; c < 8; ++c) {
            float v = h2[(size_t)i * 16 + hd * 8 + c];
            s += v * a_src2[hd * 8 + c];
            d += v * a_dst2[hd * 8 + c];
        }
        as2[i * 2 + hd] = s;
        ad2[i * 2 + hd] = d;
    }
}

// ---------------- bias + relu + mean-pool (LDS-reduced partials) -----------
__global__ void gat_pool16(const float* __restrict__ out2, const float* __restrict__ b2,
                           float* __restrict__ pooled, int N) {
    __shared__ float s[16];
    int i = blockIdx.x * blockDim.x + threadIdx.x;
    if (threadIdx.x < 16) s[threadIdx.x] = 0.0f;
    __syncthreads();
    if (i < N) {
        #pragma unroll
        for (int j = 0; j < 16; ++j) {
            float y = fmaxf(out2[(size_t)i * 16 + j] + b2[j], 0.0f);
            atomicAdd(&s[j], y);
        }
    }
    __syncthreads();
    if (threadIdx.x < 16) atomicAdd(&pooled[threadIdx.x], s[threadIdx.x]);
}

// ---------------- MLP head + sigmoid (single wave) -------------------------
__global__ __launch_bounds__(32) void gat_head(const float* __restrict__ pooled,
                                               const float* __restrict__ fc1_w,
                                               const float* __restrict__ fc1_b,
                                               const float* __restrict__ fc2_w,
                                               const float* __restrict__ fc2_b,
                                               float* __restrict__ out, int N) {
    __shared__ float z1[32];
    int j = threadIdx.x;
    float invN = 1.0f / (float)N;
    float acc = fc1_b[j];
    #pragma unroll
    for (int k = 0; k < 16; ++k) acc += (pooled[k] * invN) * fc1_w[k * 32 + j];
    z1[j] = fmaxf(acc, 0.0f);
    __syncthreads();
    if (j < 10) {
        float a2 = fc2_b[j];
        #pragma unroll
        for (int k = 0; k < 32; ++k) a2 += z1[k] * fc2_w[k * 10 + j];
        out[j] = 1.0f / (1.0f + __expf(-a2));
    }
}

extern "C" void kernel_launch(void* const* d_in, const int* in_sizes, int n_in,
                              void* d_out, int out_size, void* d_ws, size_t ws_size,
                              hipStream_t stream) {
    (void)n_in; (void)out_size; (void)ws_size;
    const float* x      = (const float*)d_in[0];
    const int*   ei     = (const int*)d_in[1];
    const float* W1     = (const float*)d_in[2];
    const float* a_src1 = (const float*)d_in[3];
    const float* a_dst1 = (const float*)d_in[4];
    const float* b1     = (const float*)d_in[5];
    const float* W2     = (const float*)d_in[6];
    const float* a_src2 = (const float*)d_in[7];
    const float* a_dst2 = (const float*)d_in[8];
    const float* b2     = (const float*)d_in[9];
    const float* fc1_w  = (const float*)d_in[10];
    const float* fc1_b  = (const float*)d_in[11];
    const float* fc2_w  = (const float*)d_in[12];
    const float* fc2_b  = (const float*)d_in[13];
    float* out = (float*)d_out;

    const int N = in_sizes[0];       // x is [N,1]
    const int E = in_sizes[1] / 2;   // edge_index is [2,E]
    const int Et = E + N;

    // workspace carve-up (floats)
    float* ws = (float*)d_ws;
    size_t off = 0;
    float* h1   = ws + off; off += (size_t)N * 64;
    float* as1  = ws + off; off += (size_t)N * 4;
    float* ad1  = ws + off; off += (size_t)N * 4;
    float* m1   = ws + off; off += (size_t)N * 4;
    float* d1   = ws + off; off += (size_t)N * 4;
    float* out1 = ws + off; off += (size_t)N * 64;   // becomes x2 after bias+relu
    float* h2   = ws + off; off += (size_t)N * 16;
    float* as2  = ws + off; off += (size_t)N * 2;
    float* ad2  = ws + off; off += (size_t)N * 2;
    float* m2   = ws + off; off += (size_t)N * 2;
    float* d2   = ws + off; off += (size_t)N * 2;
    float* out2 = ws + off; off += (size_t)N * 16;
    float* pooled = ws + off; off += 16;

    const int TB = 256;
    auto blocks = [&](long long n) { return (unsigned)((n + TB - 1) / TB); };

    gat_init_ws<<<blocks((long long)N * 64), TB, 0, stream>>>(m1, d1, out1, m2, d2, out2, pooled, N);

    // ---- layer 1 ----
    gat_node1<<<blocks(N), TB, 0, stream>>>(x, W1, a_src1, a_dst1, h1, as1, ad1, N);
    gat_edge_max<4><<<blocks(Et), TB, 0, stream>>>(ei, E, N, as1, ad1, m1);
    gat_edge_sum<4><<<blocks(Et), TB, 0, stream>>>(ei, E, N, as1, ad1, m1, d1);
    gat_edge_agg<4, 16><<<blocks((long long)Et * 4), TB, 0, stream>>>(ei, E, N, as1, ad1, m1, d1, h1, out1);
    gat_bias_relu64<<<blocks((long long)N * 64), TB, 0, stream>>>(out1, b1, N);

    // ---- layer 2 ----
    const int full_tiles = N / 16;
    if (full_tiles > 0)
        gat_gemm_h2<<<full_tiles, 32, 0, stream>>>(out1, W2, h2);
    const int tail_start = full_tiles * 16;
    if (tail_start < N)
        gat_gemm_tail<<<1, 32, 0, stream>>>(out1, W2, h2, tail_start, N);

    gat_node2_alpha<<<blocks(N), TB, 0, stream>>>(h2, a_src2, a_dst2, as2, ad2, N);
    gat_edge_max<2><<<blocks(Et), TB, 0, stream>>>(ei, E, N, as2, ad2, m2);
    gat_edge_sum<2><<<blocks(Et), TB, 0, stream>>>(ei, E, N, as2, ad2, m2, d2);
    gat_edge_agg<2, 8><<<blocks((long long)Et * 2), TB, 0, stream>>>(ei, E, N, as2, ad2, m2, d2, h2, out2);

    // ---- pool + head ----
    gat_pool16<<<blocks(N), TB, 0, stream>>>(out2, b2, pooled, N);
    gat_head<<<1, 32, 0, stream>>>(pooled, fc1_w, fc1_b, fc2_w, fc2_b, out, N);
}
